// Lumiere1M_88347477279183
// MI455X (gfx1250) — compile-verified, matrix operand order checked by reading, not run on hardware
//
#include <hip/hip_runtime.h>
#include <math.h>

typedef unsigned short u16;
typedef unsigned int   u32;
typedef __attribute__((ext_vector_type(16))) __bf16        v16bf;
typedef __attribute__((ext_vector_type(8)))  float         v8f;
typedef __attribute__((ext_vector_type(8)))  unsigned int  v8u;

__device__ __forceinline__ u16 f2bf(float f) {
  u32 u = __builtin_bit_cast(u32, f);
  u += 0x7FFFu + ((u >> 16) & 1u);   // round-to-nearest-even
  return (u16)(u >> 16);
}

// CDNA5 async global->LDS staging (ASYNCcnt-tracked, ISA §10 / §15.18.3).
__device__ __forceinline__ void async_ld128(unsigned lds_off, const u16* g) {
  asm volatile("global_load_async_to_lds_b128 %0, %1, off"
               :: "v"(lds_off), "v"((unsigned long long)g)
               : "memory");
}
template <int N>
__device__ __forceinline__ void wait_async() {
  asm volatile("s_wait_asynccnt %0" :: "i"(N) : "memory");
}

// ---------------------------------------------------------------------------
// bf16 GEMM, fp32 accumulate, V_WMMA_F32_16X16X32_BF16, double-buffered async
// global->LDS pipeline.
//   C[M,N] = A[M,K] * B^T  where B is stored N x K row-major (ldb = n-row stride)
// 256 threads = 8 waves (4x2). Block tile 128 x BN. Wave tile 32 x (BN/2).
// BN=128: 8 WMMA / wave / k-step.  BN=64: 4 WMMA (for N=64 GEMMs).
// M % 128 == 0, N % BN == 0, K % 64 == 0 (even number of 32-wide k-steps).
// ---------------------------------------------------------------------------
template <int BN, bool OUTBF>
__global__ __launch_bounds__(256) void gemm_bf16_wmma(
    const u16* __restrict__ A, long lda,
    const u16* __restrict__ B, long ldb,
    void* __restrict__ Cp, long ldc, int K)
{
  constexpr int WN = BN / 32;              // 16-wide n-subtiles per wave (4 or 2)
  constexpr int NL = (BN == 128) ? 4 : 3;  // async b128 loads per thread per tile
  __shared__ u16 As[2][128 * 32];          // [buf][m][k]
  __shared__ u16 Bs[2][BN * 32];           // [buf][n][k]
  const int tid  = threadIdx.x;
  const int lane = tid & 31, wave = tid >> 5;
  const int wr = wave >> 1, wc = wave & 1;
  const int lo = lane & 15, kh = lane >> 4;
  const long m0 = (long)blockIdx.y * 128;
  const long n0 = (long)blockIdx.x * BN;

  const int lrow = tid >> 2, cg = tid & 3;           // 64 rows x 16B per pass
  const u16* aptr = A + (m0 + lrow) * lda + cg * 8;
  const u16* bptr = B + (n0 + lrow) * ldb + cg * 8;
  const unsigned as0 = (unsigned)(unsigned long long)&As[0][lrow * 32 + cg * 8];
  const unsigned bs0 = (unsigned)(unsigned long long)&Bs[0][lrow * 32 + cg * 8];

  v8f acc[2][WN];
#pragma unroll
  for (int i = 0; i < 2; ++i)
#pragma unroll
    for (int j = 0; j < WN; ++j) acc[i][j] = (v8f){};

  auto stage = [&](int buf, int k0) {
    unsigned ao = as0 + (unsigned)buf * (128 * 32 * 2);
    unsigned bo = bs0 + (unsigned)buf * (BN * 32 * 2);
    async_ld128(ao,           aptr + k0);
    async_ld128(ao + 64 * 64, aptr + 64 * lda + k0);
    async_ld128(bo,           bptr + k0);
    if constexpr (BN == 128)
      async_ld128(bo + 64 * 64, bptr + 64 * ldb + k0);
  };

  auto compute = [&](const u16* Asb, const u16* Bsb) {
    const u32* Asu = (const u32*)Asb;
    const u32* Bsu = (const u32*)Bsb;
    v16bf af[2], bfm[WN];
#pragma unroll
    for (int i = 0; i < 2; ++i) {          // A frags: 16-bit A 16x32 layout
      v8u a;
      int r = wr * 32 + i * 16 + lo;
#pragma unroll
      for (int v = 0; v < 8; ++v) {
        int kk = ((v < 4) ? v : v + 4) + 4 * kh;
        a[v] = Asu[r * 16 + kk];
      }
      af[i] = __builtin_bit_cast(v16bf, a);
    }
#pragma unroll
    for (int j = 0; j < WN; ++j) {         // B frags (n across lanes, mirrored)
      v8u b;
      int c = wc * (16 * WN) + j * 16 + lo;
#pragma unroll
      for (int v = 0; v < 8; ++v) {
        int kk = ((v < 4) ? v : v + 4) + 4 * kh;
        b[v] = Bsu[c * 16 + kk];
      }
      bfm[j] = __builtin_bit_cast(v16bf, b);
    }
#pragma unroll
    for (int i = 0; i < 2; ++i)
#pragma unroll
      for (int j = 0; j < WN; ++j)
        acc[i][j] = __builtin_amdgcn_wmma_f32_16x16x32_bf16(
            false, af[i], false, bfm[j], (short)0, acc[i][j], false, false);
  };

  // Double-buffered pipeline: T = K/32 is even for all GEMMs in this model.
  const int T = K >> 5;
  stage(0, 0);
  for (int t = 0; t < T; t += 2) {
    stage(1, (t + 1) * 32);                // prefetch odd tile into buf1
    wait_async<NL>();                      // tile t (buf0) loads complete
    __syncthreads();
    compute(As[0], Bs[0]);
    __syncthreads();
    if (t + 2 < T) { stage(0, (t + 2) * 32); wait_async<NL>(); }
    else           { wait_async<0>(); }
    __syncthreads();
    compute(As[1], Bs[1]);
    __syncthreads();
  }

  // Epilogue per documented f32 D layout (lanes 0-15 rows 0-7, 16-31 rows 8-15).
#pragma unroll
  for (int i = 0; i < 2; ++i)
#pragma unroll
    for (int j = 0; j < WN; ++j) {
      long col = n0 + wc * (16 * WN) + j * 16 + lo;
#pragma unroll
      for (int v = 0; v < 8; ++v) {
        long row = m0 + wr * 32 + i * 16 + kh * 8 + v;
        float val = acc[i][j][v];
        if constexpr (OUTBF) ((u16*)Cp)[row * ldc + col] = f2bf(val);
        else                 ((float*)Cp)[row * ldc + col] = val;
      }
    }
}

// ---------------------------------------------------------------------------
// fp32 [R,C] -> bf16 [C,R] (weight prep: transpose + downconvert, coalesced).
__global__ void cvt_transpose(const float* __restrict__ in, u16* __restrict__ out,
                              int R, int C) {
  __shared__ float t[32][33];
  int c0 = blockIdx.x * 32, r0 = blockIdx.y * 32;
  int tx = threadIdx.x, ty = threadIdx.y;            // 32 x 8
#pragma unroll
  for (int u = 0; u < 4; ++u)
    t[ty + u * 8][tx] = in[(long)(r0 + ty + u * 8) * C + c0 + tx];
  __syncthreads();
#pragma unroll
  for (int u = 0; u < 4; ++u)
    out[(long)(c0 + ty + u * 8) * R + r0 + tx] = f2bf(t[tx][ty + u * 8]);
}

// bf16 [R,C] -> bf16 [C,R] (V transpose per batch).
__global__ void transpose_bf16(const u16* __restrict__ in, u16* __restrict__ out,
                               int R, int C) {
  __shared__ u16 t[32][33];
  int c0 = blockIdx.x * 32, r0 = blockIdx.y * 32;
  int tx = threadIdx.x, ty = threadIdx.y;            // 32 x 8
#pragma unroll
  for (int u = 0; u < 4; ++u)
    t[ty + u * 8][tx] = in[(long)(r0 + ty + u * 8) * C + c0 + tx];
  __syncthreads();
#pragma unroll
  for (int u = 0; u < 4; ++u)
    out[(long)(c0 + ty + u * 8) * R + r0 + tx] = t[tx][ty + u * 8];
}

__global__ void embed_pe(const int* __restrict__ x, const unsigned char* __restrict__ pm,
                         const float* __restrict__ emb, float* __restrict__ h,
                         u16* __restrict__ hb) {
  const int D = 512, S = 2048;
  long tok = blockIdx.x;
  int pos = (int)(tok % S);
  int id  = x[tok];
  bool p  = pm[tok] != 0;
  int t = threadIdx.x;                       // 256 threads, 2 dims each
#pragma unroll
  for (int u = 0; u < 2; ++u) {
    int d = t + u * 256;
    int i = d >> 1;
    float ang = (float)pos * __expf(-(2.0f * (float)i / (float)D) * 9.210340371976184f);
    float pe  = p ? 0.0f : ((d & 1) ? __cosf(ang) : __sinf(ang));
    float v   = emb[(long)id * D + d] + pe;
    h [tok * (long)D + d] = v;
    hb[tok * (long)D + d] = f2bf(v);
  }
}

// Stable masked softmax per row (causal mask INCLUDES diagonal; max clamped >=0;
// denom + 1e-9 so fully-masked rows -> zeros). Emits bf16 attention row.
__global__ void attn_softmax(const float* __restrict__ sc, u16* __restrict__ attn,
                             const unsigned char* __restrict__ pm) {
  const int S = 2048;
  int i = blockIdx.x, t = threadIdx.x;       // 256 threads, 8 cols each
  __shared__ float red[256];
  bool qp = pm[i] != 0;
  float e8[8];
  float m = -3.0e38f;
#pragma unroll
  for (int u = 0; u < 8; ++u) {
    int j = t + u * 256;
    bool msk = (j >= i) || qp || (pm[j] != 0);
    float v = msk ? -__builtin_inff() : sc[(long)i * S + j] * 0.125f;
    e8[u] = v;
    m = fmaxf(m, v);
  }
  red[t] = m; __syncthreads();
  for (int o = 128; o; o >>= 1) { if (t < o) red[t] = fmaxf(red[t], red[t + o]); __syncthreads(); }
  m = fmaxf(red[0], 0.0f);
  __syncthreads();
  float s = 0.f;
#pragma unroll
  for (int u = 0; u < 8; ++u) {
    float e = (e8[u] == -__builtin_inff()) ? 0.0f : __expf(e8[u] - m);
    e8[u] = e; s += e;
  }
  red[t] = s; __syncthreads();
  for (int o = 128; o; o >>= 1) { if (t < o) red[t] += red[t + o]; __syncthreads(); }
  float inv = 1.0f / (red[0] + 1e-9f);
#pragma unroll
  for (int u = 0; u < 8; ++u)
    attn[(long)i * S + t + u * 256] = f2bf(e8[u] * inv);
}

__global__ void bias_gelu(const float* __restrict__ in, const float* __restrict__ bias,
                          u16* __restrict__ out, int N, long total) {
  long i = (long)blockIdx.x * blockDim.x + threadIdx.x;
  long st = (long)gridDim.x * blockDim.x;
  for (; i < total; i += st) {
    float xv = in[i] + bias[i % N];
    float g = 0.5f * xv * (1.0f + erff(xv * 0.7071067811865475f));   // exact GELU
    out[i] = f2bf(g);
  }
}

// h = LN(h + delta (+bias)) * gs + gb ; also emits bf16 copy of h.
__global__ void residual_ln(float* __restrict__ h, u16* __restrict__ hb,
                            const float* __restrict__ delta, const float* __restrict__ bias,
                            const float* __restrict__ gs, const float* __restrict__ gb) {
  const int D = 512;
  long r = blockIdx.x;
  int t = threadIdx.x;                       // 128 threads, 4 dims each
  __shared__ float red[128];
  float xv[4]; float s = 0.f;
#pragma unroll
  for (int u = 0; u < 4; ++u) {
    int d = t + u * 128;
    float v = h[r * D + d] + delta[r * D + d] + (bias ? bias[d] : 0.0f);
    xv[u] = v; s += v;
  }
  red[t] = s; __syncthreads();
  for (int o = 64; o; o >>= 1) { if (t < o) red[t] += red[t + o]; __syncthreads(); }
  float mu = red[0] / (float)D;
  __syncthreads();
  float s2 = 0.f;
#pragma unroll
  for (int u = 0; u < 4; ++u) { float dd = xv[u] - mu; s2 += dd * dd; }
  red[t] = s2; __syncthreads();
  for (int o = 64; o; o >>= 1) { if (t < o) red[t] += red[t + o]; __syncthreads(); }
  float inv = rsqrtf(red[0] / (float)D + 1e-5f);
#pragma unroll
  for (int u = 0; u < 4; ++u) {
    int d = t + u * 128;
    float y = (xv[u] - mu) * inv * gs[d] + gb[d];
    h [r * D + d] = y;
    hb[r * D + d] = f2bf(y);
  }
}

// ---------------------------------------------------------------------------
extern "C" void kernel_launch(void* const* d_in, const int* in_sizes, int n_in,
                              void* d_out, int out_size, void* d_ws, size_t ws_size,
                              hipStream_t stream) {
  (void)in_sizes; (void)n_in; (void)out_size; (void)ws_size;
  const int V = 32000, S = 2048, D = 512, H = 8, DK = 64, DFF = 2048, L = 2, Bn = 2;
  const long BS = (long)Bn * S;

  const int*           x  = (const int*)d_in[0];
  const unsigned char* pm = (const unsigned char*)d_in[1];
  const float* emb  = (const float*)d_in[2];
  const float* wq   = (const float*)d_in[3];
  const float* wk   = (const float*)d_in[4];
  const float* wv   = (const float*)d_in[5];
  const float* wo   = (const float*)d_in[6];
  const float* ln1s = (const float*)d_in[7];
  const float* ln1b = (const float*)d_in[8];
  const float* ln2s = (const float*)d_in[9];
  const float* ln2b = (const float*)d_in[10];
  const float* w1   = (const float*)d_in[11];
  const float* b1   = (const float*)d_in[12];
  const float* w2   = (const float*)d_in[13];
  const float* b2   = (const float*)d_in[14];
  const float* wout = (const float*)d_in[15];
  float* out = (float*)d_out;

  char* base = (char*)d_ws; size_t off = 0;
  auto alloc = [&](size_t bytes) -> char* {
    char* p = base + off; off += (bytes + 255) & ~(size_t)255; return p;
  };
  float* h    = (float*)alloc(BS * D * 4);
  u16*   hb   = (u16*)  alloc(BS * D * 2);
  u16*   qb   = (u16*)  alloc(BS * D * 2);
  u16*   kb   = (u16*)  alloc(BS * D * 2);
  u16*   vb   = (u16*)  alloc(BS * D * 2);
  u16*   vtb  = (u16*)  alloc(BS * D * 2);          // [Bn*D, S] transposed V
  u16*   ob   = (u16*)  alloc(BS * D * 2);
  u16*   ffb  = (u16*)  alloc(BS * (long)DFF * 2);
  float* tmpf = (float*)alloc(BS * (long)DFF * 4);  // reused fp32 scratch
  float* sc   = (float*)alloc((long)S * S * 4);     // one head's scores
  u16*   attn = (u16*)  alloc((long)S * S * 2);
  u16*   wqT  = (u16*)  alloc((long)L * D * D * 2);       // [N=D, K=D]
  u16*   wkT  = (u16*)  alloc((long)L * D * D * 2);
  u16*   wvT  = (u16*)  alloc((long)L * D * D * 2);
  u16*   woT  = (u16*)  alloc((long)L * D * D * 2);
  u16*   w1T  = (u16*)  alloc((long)L * D * DFF * 2);     // [DFF, D]
  u16*   w2T  = (u16*)  alloc((long)L * D * DFF * 2);     // [D, DFF]
  u16*   woutT= (u16*)  alloc((long)D * V * 2);           // [V, D]

  dim3 tb(32, 8);
  auto ctr = [&](const float* s_, u16* d_, int R, int C) {  // fp32 RxC -> bf16 CxR
    cvt_transpose<<<dim3(C / 32, R / 32), tb, 0, stream>>>(s_, d_, R, C);
  };
  for (int l = 0; l < L; ++l) {
    ctr(wq + (long)l * D * D,  wqT + (long)l * D * D,  D, D);
    ctr(wk + (long)l * D * D,  wkT + (long)l * D * D,  D, D);
    ctr(wv + (long)l * D * D,  wvT + (long)l * D * D,  D, D);
    ctr(wo + (long)l * D * D,  woT + (long)l * D * D,  D, D);
    ctr(w1 + (long)l * D * DFF, w1T + (long)l * D * DFF, D, DFF);
    ctr(w2 + (long)l * D * DFF, w2T + (long)l * D * DFF, DFF, D);
  }
  ctr(wout, woutT, D, V);

  embed_pe<<<(int)BS, 256, 0, stream>>>(x, pm, emb, h, hb);

  auto gemmF32 = [&](const u16* A, long lda, const u16* B, long ldb,
                     float* C, long ldc, int M, int N, int K) {
    gemm_bf16_wmma<128, false><<<dim3(N / 128, M / 128), 256, 0, stream>>>(
        A, lda, B, ldb, (void*)C, ldc, K);
  };
  auto gemmBF = [&](const u16* A, long lda, const u16* B, long ldb,
                    u16* C, long ldc, int M, int N, int K) {
    gemm_bf16_wmma<128, true><<<dim3(N / 128, M / 128), 256, 0, stream>>>(
        A, lda, B, ldb, (void*)C, ldc, K);
  };
  auto gemmN64 = [&](const u16* A, long lda, const u16* B, long ldb,
                     u16* C, long ldc, int M, int N, int K) {
    gemm_bf16_wmma<64, true><<<dim3(N / 64, M / 128), 256, 0, stream>>>(
        A, lda, B, ldb, (void*)C, ldc, K);
  };

  for (int l = 0; l < L; ++l) {
    gemmBF(hb, D, wqT + (long)l * D * D, D, qb, D, (int)BS, D, D);
    gemmBF(hb, D, wkT + (long)l * D * D, D, kb, D, (int)BS, D, D);
    gemmBF(hb, D, wvT + (long)l * D * D, D, vb, D, (int)BS, D, D);

    for (int b = 0; b < Bn; ++b)   // V -> V^T per batch: [S,D] -> [D,S]
      transpose_bf16<<<dim3(D / 32, S / 32), tb, 0, stream>>>(
          vb + (long)b * S * D, vtb + (long)b * D * S, S, D);

    for (int b = 0; b < Bn; ++b)
      for (int hd = 0; hd < H; ++hd) {
        long hoff = (long)b * S * D + (long)hd * DK;
        // scores = Q * K^T   (K is naturally N x K row-major, ldb = D)
        gemmF32(qb + hoff, D, kb + hoff, D, sc, S, S, S, DK);
        attn_softmax<<<S, 256, 0, stream>>>(sc, attn, pm + (long)b * S);
        // o_head = attn * V  (B = V^T rows of head hd, ldb = S)
        gemmN64(attn, S, vtb + ((long)b * D + (long)hd * DK) * S, S,
                ob + hoff, D, S, DK, S);
      }

    gemmF32(ob, D, woT + (long)l * D * D, D, tmpf, D, (int)BS, D, D);
    residual_ln<<<(int)BS, 128, 0, stream>>>(h, hb, tmpf, nullptr,
                                             ln1s + (long)l * D, ln1b + (long)l * D);
    gemmF32(hb, D, w1T + (long)l * D * DFF, D, tmpf, DFF, (int)BS, DFF, D);
    bias_gelu<<<4096, 256, 0, stream>>>(tmpf, b1 + (long)l * DFF, ffb, DFF, BS * (long)DFF);
    gemmF32(ffb, DFF, w2T + (long)l * D * DFF, DFF, tmpf, D, (int)BS, D, DFF);
    residual_ln<<<(int)BS, 128, 0, stream>>>(h, hb, tmpf, b2 + (long)l * D,
                                             ln2s + (long)l * D, ln2b + (long)l * D);
  }

  // logits = h @ w_out   (dominant GEMM: 4096 x 32000 x 512)
  gemmF32(hb, D, woutT, D, out, V, (int)BS, V, D);
}
